// DRSformer_47261820125508
// MI455X (gfx1250) — matrix operand order repaired and checked by reading, not compiled, not run on hardware
//
#include <hip/hip_runtime.h>
#include <hip/hip_bf16.h>

typedef __attribute__((ext_vector_type(16))) _Float16 v16h;
typedef __attribute__((ext_vector_type(8)))  float    v8f;

#define NTOT 16384   // H*W
#define CH3  1152    // 3*DIM
#define DIMC 384
#define NHEAD 8
#define HC   48      // DIM/HEADS

#define LDK 48       // LDS row stride in halfs (96B: 16B-aligned, conflict-free rotation)

// ---------------------------------------------------------------------------
// Generic tiled WMMA GEMM:  C[b][m][n] = sum_k A[m][k] * B[b][k][n]
// BM=64, BN=64, BK=32, 256 threads = 8 waves, each wave owns 1 m-subtile and
// 2 n-subtiles (16x16 each). f32 in memory, f16 in LDS/WMMA, f32 accumulate.
// All global loads are b128 (float4); transpose happens on the LDS-store side.
// ---------------------------------------------------------------------------
__global__ void __launch_bounds__(256)
gemm_wmma_f16(const float* __restrict__ A, const float* __restrict__ Bm,
              float* __restrict__ C, int M, int K, int N) {
  const int n0 = blockIdx.x * 64;
  const int m0 = blockIdx.y * 64;
  const int b  = blockIdx.z;
  const float* B  = Bm + (size_t)b * K * N;
  float*       Cb = C  + (size_t)b * M * N;

  __shared__ _Float16 Ah[64][LDK];   // [m][k]
  __shared__ _Float16 Bh[64][LDK];   // [n][k] (transposed tile)

  const int tid   = threadIdx.x;
  const int lane  = tid & 31;
  const int wave  = tid >> 5;
  const int half_ = lane >> 4;
  const int l16   = lane & 15;

  const int m_sub  = wave & 3;
  const int n_sub0 = (wave >> 2) * 2;

  // A-tile mapping: 8 contiguous k per thread
  const int a_row = tid >> 2;            // 0..63
  const int a_kc  = (tid & 3) * 8;       // 0,8,16,24
  // B-tile mapping: 8 contiguous n per thread (coalesced b128 reads)
  const int b_k   = tid >> 3;            // 0..31
  const int b_nc  = (tid & 7) * 8;       // 0..56

  v8f acc0 = {0.f,0.f,0.f,0.f,0.f,0.f,0.f,0.f};
  v8f acc1 = acc0;

  for (int k0 = 0; k0 < K; k0 += 32) {
    // ---- A tile 64x32: two float4 loads, 8 contiguous half stores
    {
      const float4* g = (const float4*)(A + (size_t)(m0 + a_row) * K + (k0 + a_kc));
      const float4 u0 = g[0], u1 = g[1];
      Ah[a_row][a_kc + 0] = (_Float16)u0.x; Ah[a_row][a_kc + 1] = (_Float16)u0.y;
      Ah[a_row][a_kc + 2] = (_Float16)u0.z; Ah[a_row][a_kc + 3] = (_Float16)u0.w;
      Ah[a_row][a_kc + 4] = (_Float16)u1.x; Ah[a_row][a_kc + 5] = (_Float16)u1.y;
      Ah[a_row][a_kc + 6] = (_Float16)u1.z; Ah[a_row][a_kc + 7] = (_Float16)u1.w;
    }
    // ---- B tile 32x64: two float4 loads along n, transpose on LDS store
    {
      const float4* g = (const float4*)(B + (size_t)(k0 + b_k) * N + (n0 + b_nc));
      const float4 u0 = g[0], u1 = g[1];
      Bh[b_nc + 0][b_k] = (_Float16)u0.x; Bh[b_nc + 1][b_k] = (_Float16)u0.y;
      Bh[b_nc + 2][b_k] = (_Float16)u0.z; Bh[b_nc + 3][b_k] = (_Float16)u0.w;
      Bh[b_nc + 4][b_k] = (_Float16)u1.x; Bh[b_nc + 5][b_k] = (_Float16)u1.y;
      Bh[b_nc + 6][b_k] = (_Float16)u1.z; Bh[b_nc + 7][b_k] = (_Float16)u1.w;
    }
    if (k0 + 32 < K)  // hint next B tile toward the caches
      __builtin_prefetch(B + (size_t)(k0 + 32 + b_k) * N + n0 + b_nc, 0, 1);
    __syncthreads();

    v16h af, bf0, bf1;
    #pragma unroll
    for (int j = 0; j < 16; ++j) {
      const int kk = (j < 8 ? j : j + 8) + 8 * half_;     // A 16x32 lane layout
      af[j] = Ah[m_sub * 16 + l16][kk];
    }
    #pragma unroll
    for (int j = 0; j < 16; ++j) {
      const int kk = (j < 8 ? j : j + 8) + 8 * half_;     // B 32x16 lane layout
      bf0[j] = Bh[n_sub0 * 16 + l16][kk];
      bf1[j] = Bh[(n_sub0 + 1) * 16 + l16][kk];
    }
    acc0 = __builtin_amdgcn_wmma_f32_16x16x32_f16(false, af, false, bf0, (short)0, acc0, false, false);
    acc1 = __builtin_amdgcn_wmma_f32_16x16x32_f16(false, af, false, bf1, (short)0, acc1, false, false);
    __syncthreads();
  }

  #pragma unroll
  for (int r = 0; r < 8; ++r) {                 // D: VGPR r -> M = r + 8*half
    const int mg = m0 + m_sub * 16 + r + 8 * half_;
    const int ng = n0 + n_sub0 * 16 + l16;
    Cb[(size_t)mg * N + ng]      = acc0[r];
    Cb[(size_t)mg * N + ng + 16] = acc1[r];
  }
}

// ---------------------------------------------------------------------------
// Depthwise 3x3, SAME zero padding. [B][CH][128][128]
// ---------------------------------------------------------------------------
__global__ void __launch_bounds__(256)
dwconv3x3(const float* __restrict__ in, const float* __restrict__ w,
          float* __restrict__ out, long total, int CH) {
  long idx = (long)blockIdx.x * 256 + threadIdx.x;
  if (idx >= total) return;
  const int x  = (int)(idx & 127);
  const int y  = (int)((idx >> 7) & 127);
  const int ch = (int)((idx >> 14) % CH);
  const long base = (idx >> 14) << 14;          // (b*CH + ch) * 16384
  const float* wp = w + ch * 9;
  float s = 0.f;
  #pragma unroll
  for (int ky = -1; ky <= 1; ++ky) {
    const int yy = y + ky;
    if (yy < 0 || yy > 127) continue;
    #pragma unroll
    for (int kx = -1; kx <= 1; ++kx) {
      const int xx = x + kx;
      if (xx < 0 || xx > 127) continue;
      s += wp[(ky + 1) * 3 + (kx + 1)] * in[base + yy * 128 + xx];
    }
  }
  out[idx] = s;
}

// ---------------------------------------------------------------------------
// Per-row inverse L2 norm over N for the q (c in [0,384)) and k (c in [384,768))
// channel rows of qkv. scale[b*768 + c] = 1/max(||row||, 1e-12)
// ---------------------------------------------------------------------------
__global__ void __launch_bounds__(256)
l2norm_rows(const float* __restrict__ qkv, float* __restrict__ scale) {
  const int rid = blockIdx.x;                    // 0 .. B*768-1
  const int b = rid / 768, c = rid % 768;
  const float4* p = (const float4*)(qkv + ((size_t)b * CH3 + c) * NTOT);
  float s = 0.f;
  for (int i = threadIdx.x; i < NTOT / 4; i += 256) {
    const float4 v = p[i];
    s += v.x * v.x + v.y * v.y + v.z * v.z + v.w * v.w;
  }
  __shared__ float red[256];
  red[threadIdx.x] = s; __syncthreads();
  for (int off = 128; off > 0; off >>= 1) {
    if (threadIdx.x < off) red[threadIdx.x] += red[threadIdx.x + off];
    __syncthreads();
  }
  if (threadIdx.x == 0) scale[rid] = 1.f / fmaxf(sqrtf(red[0]), 1e-12f);
}

// ---------------------------------------------------------------------------
// attn[b,h,c,d] = temp[h]*sq[c]*sk[d] * sum_n q[c,n]*k[d,n]
// One block per 16x16 tile (3x3 tiles) per (b,h). 8 waves split K=16384,
// partial WMMA accumulators reduced through LDS. All global reads are b128.
// ---------------------------------------------------------------------------
__global__ void __launch_bounds__(256)
attn_gram(const float* __restrict__ qkv, const float* __restrict__ scale,
          const float* __restrict__ temperature, float* __restrict__ attn) {
  const int tile = blockIdx.x, mt = tile / 3, nt = tile % 3;
  const int bh = blockIdx.y, b = bh >> 3, h = bh & 7;
  const int lane = threadIdx.x & 31, wave = threadIdx.x >> 5;
  const int half_ = lane >> 4, l16 = lane & 15;

  const float* qp = qkv + ((size_t)b * CH3 + h * HC + mt * 16 + l16) * NTOT;
  const float* kp = qkv + ((size_t)b * CH3 + DIMC + h * HC + nt * 16 + l16) * NTOT;

  v8f acc = {0.f,0.f,0.f,0.f,0.f,0.f,0.f,0.f};
  const int kbeg = wave * 2048, kend = kbeg + 2048;
  for (int k0 = kbeg; k0 < kend; k0 += 32) {
    // A chunk: k = 8*half + {0..7} and 16+8*half + {0..7}   (16B aligned)
    const float4* qa = (const float4*)(qp + k0 + 8 * half_);
    const float4 a0 = qa[0], a1 = qa[1];         // j = 0..7
    const float4 a2 = qa[4], a3 = qa[5];         // j = 8..15 (offset +16 floats)
    // B chunk: k = 8*half + {0..7} and 16+8*half + {0..7}   (16B aligned)
    const float4* kb = (const float4*)(kp + k0 + 8 * half_);
    const float4 b0 = kb[0], b1 = kb[1], b2 = kb[4], b3 = kb[5];

    v16h af, bf;
    af[0]=(_Float16)a0.x; af[1]=(_Float16)a0.y; af[2]=(_Float16)a0.z; af[3]=(_Float16)a0.w;
    af[4]=(_Float16)a1.x; af[5]=(_Float16)a1.y; af[6]=(_Float16)a1.z; af[7]=(_Float16)a1.w;
    af[8]=(_Float16)a2.x; af[9]=(_Float16)a2.y; af[10]=(_Float16)a2.z; af[11]=(_Float16)a2.w;
    af[12]=(_Float16)a3.x; af[13]=(_Float16)a3.y; af[14]=(_Float16)a3.z; af[15]=(_Float16)a3.w;
    bf[0]=(_Float16)b0.x; bf[1]=(_Float16)b0.y; bf[2]=(_Float16)b0.z; bf[3]=(_Float16)b0.w;
    bf[4]=(_Float16)b1.x; bf[5]=(_Float16)b1.y; bf[6]=(_Float16)b1.z; bf[7]=(_Float16)b1.w;
    bf[8]=(_Float16)b2.x; bf[9]=(_Float16)b2.y; bf[10]=(_Float16)b2.z; bf[11]=(_Float16)b2.w;
    bf[12]=(_Float16)b3.x; bf[13]=(_Float16)b3.y; bf[14]=(_Float16)b3.z; bf[15]=(_Float16)b3.w;
    acc = __builtin_amdgcn_wmma_f32_16x16x32_f16(false, af, false, bf, (short)0, acc, false, false);
  }

  __shared__ float red[8 * 256];
  #pragma unroll
  for (int r = 0; r < 8; ++r)
    red[wave * 256 + (r + 8 * half_) * 16 + l16] = acc[r];
  __syncthreads();

  const int t = threadIdx.x;                    // element: m = t/16, d = t%16
  float s = 0.f;
  #pragma unroll
  for (int w2 = 0; w2 < 8; ++w2) s += red[w2 * 256 + t];
  const int m = t >> 4, d = t & 15;
  const int gm = mt * 16 + m, gd = nt * 16 + d;
  const float sq = scale[b * 768 + h * HC + gm];
  const float sk = scale[b * 768 + DIMC + h * HC + gd];
  attn[(((size_t)b * NHEAD + h) * HC + gm) * HC + gd] = s * sq * sk * temperature[h];
}

// ---------------------------------------------------------------------------
// Fuse the four top-k softmaxes: P = a1*p24 + a2*p32 + a3*p36 + a4*p38.
// One thread per 48-wide row; k-th largest found by counting (matches
// vals[k-1] + (attn >= thresh) semantics exactly, including ties).
// ---------------------------------------------------------------------------
__global__ void __launch_bounds__(256)
topk_combine(const float* __restrict__ attn, float* __restrict__ P,
             const float* __restrict__ a1, const float* __restrict__ a2,
             const float* __restrict__ a3, const float* __restrict__ a4) {
  const int row = blockIdx.x * 256 + threadIdx.x;
  if (row >= 2 * NHEAD * HC) return;
  const float* ar = attn + (size_t)row * HC;
  float x[HC], ex[HC], outw[HC];
  float mx = -__builtin_inff();
  for (int i = 0; i < HC; ++i) { x[i] = ar[i]; mx = fmaxf(mx, x[i]); }
  for (int i = 0; i < HC; ++i) { ex[i] = __expf(x[i] - mx); outw[i] = 0.f; }
  const int   ks[4] = {HC / 2, HC * 2 / 3, HC * 3 / 4, HC * 4 / 5}; // 24,32,36,38
  const float as[4] = {a1[0], a2[0], a3[0], a4[0]};
  for (int kk = 0; kk < 4; ++kk) {
    const int k = ks[kk];
    float thr = -__builtin_inff();
    for (int i = 0; i < HC; ++i) {
      int cnt = 0;
      for (int j = 0; j < HC; ++j) cnt += (x[j] >= x[i]) ? 1 : 0;
      if (cnt >= k && x[i] > thr) thr = x[i];
    }
    float denom = 0.f;
    for (int j = 0; j < HC; ++j) if (x[j] >= thr) denom += ex[j];
    const float ainv = as[kk] / denom;
    for (int j = 0; j < HC; ++j) if (x[j] >= thr) outw[j] += ex[j] * ainv;
  }
  for (int j = 0; j < HC; ++j) P[(size_t)row * HC + j] = outw[j];
}

// ---------------------------------------------------------------------------
// out[b, h*48+c, n] = sum_d P[b,h,c,d] * v[b,h,d,n]   (K=48 zero-padded to 64)
// Block: 128-wide n tile per (b,h); 8 waves = 8 n-subtiles, 3 m-tiles each.
// V tile loaded as float4 along n (b128 coalesced), transposed on LDS store.
// ---------------------------------------------------------------------------
__global__ void __launch_bounds__(256)
attn_apply(const float* __restrict__ P, const float* __restrict__ qkv,
           float* __restrict__ out) {
  const int n0 = blockIdx.x * 128;
  const int bh = blockIdx.y, b = bh >> 3, h = bh & 7;
  __shared__ _Float16 Ph[48][64];    // [c][d], d padded 48->64 with zeros
  __shared__ _Float16 Vh[128][72];   // [n][d], d padded, row-padded for banks
  const int tid = threadIdx.x;

  const float* Pp = P + (size_t)bh * HC * HC;
  for (int i = tid; i < 48 * 64; i += 256) {
    const int c = i >> 6, d = i & 63;
    Ph[c][d] = (d < HC) ? (_Float16)Pp[c * HC + d] : (_Float16)0.f;
  }
  // V tile: 48 rows x 128 n, float4 per thread along n (coalesced b128)
  const float* vp = qkv + ((size_t)b * CH3 + 2 * DIMC + h * HC) * NTOT;
  #pragma unroll
  for (int e = 0; e < 6; ++e) {                  // 48*32 quads / 256 threads
    const int idx = tid + 256 * e;               // 0..1535
    const int k  = idx >> 5;                     // 0..47
    const int n4 = (idx & 31) * 4;               // 0..124
    const float4 u = *(const float4*)(vp + (size_t)k * NTOT + n0 + n4);
    Vh[n4 + 0][k] = (_Float16)u.x; Vh[n4 + 1][k] = (_Float16)u.y;
    Vh[n4 + 2][k] = (_Float16)u.z; Vh[n4 + 3][k] = (_Float16)u.w;
  }
  // zero-fill padded d = 48..63
  #pragma unroll
  for (int e = 0; e < 8; ++e) {
    const int idx = tid + 256 * e;               // 0..2047
    const int n = idx & 127, k = 48 + (idx >> 7);
    Vh[n][k] = (_Float16)0.f;
  }
  __syncthreads();

  const int lane = tid & 31, wave = tid >> 5;
  const int half_ = lane >> 4, l16 = lane & 15;
  v8f acc[3];
  #pragma unroll
  for (int m = 0; m < 3; ++m) acc[m] = (v8f){0.f,0.f,0.f,0.f,0.f,0.f,0.f,0.f};

  #pragma unroll
  for (int ksplit = 0; ksplit < 2; ++ksplit) {
    const int kb = ksplit * 32;
    v16h bf;
    #pragma unroll
    for (int j = 0; j < 16; ++j)
      bf[j] = Vh[wave * 16 + l16][kb + 8 * half_ + (j < 8 ? j : j + 8)];
    #pragma unroll
    for (int m = 0; m < 3; ++m) {
      v16h af;
      #pragma unroll
      for (int j = 0; j < 16; ++j)
        af[j] = Ph[m * 16 + l16][kb + 8 * half_ + (j < 8 ? j : j + 8)];
      acc[m] = __builtin_amdgcn_wmma_f32_16x16x32_f16(false, af, false, bf, (short)0, acc[m], false, false);
    }
  }

  #pragma unroll
  for (int m = 0; m < 3; ++m)
    #pragma unroll
    for (int r = 0; r < 8; ++r) {
      const int ch = h * HC + m * 16 + r + 8 * half_;
      out[((size_t)b * DIMC + ch) * NTOT + n0 + wave * 16 + l16] = acc[m][r];
    }
}

// ---------------------------------------------------------------------------
extern "C" void kernel_launch(void* const* d_in, const int* in_sizes, int n_in,
                              void* d_out, int out_size, void* d_ws, size_t ws_size,
                              hipStream_t stream) {
  const float* x      = (const float*)d_in[0];
  const float* w_qkv  = (const float*)d_in[1];
  const float* w_dw   = (const float*)d_in[2];
  const float* w_proj = (const float*)d_in[3];
  const float* temp   = (const float*)d_in[4];
  const float* a1     = (const float*)d_in[5];
  const float* a2     = (const float*)d_in[6];
  const float* a3     = (const float*)d_in[7];
  const float* a4     = (const float*)d_in[8];
  float* outp = (float*)d_out;

  // workspace layout (floats)
  float* ws       = (float*)d_ws;
  const size_t qkv_elems = (size_t)2 * CH3 * NTOT;      // 37,748,736
  float* qkv_raw  = ws;                                  // also reused as attn_out
  float* qkv_dw   = ws + qkv_elems;
  float* scale    = qkv_dw + qkv_elems;                  // 2*768
  float* attn     = scale + 2 * 768;                     // 2*8*48*48
  float* P        = attn + (size_t)2 * NHEAD * HC * HC;  // 2*8*48*48
  float* attn_out = qkv_raw;                             // reuse (qkv_raw dead after dwconv)

  // 1) qkv = w_qkv @ x          [1152x384]x[384x16384] per batch
  gemm_wmma_f16<<<dim3(NTOT / 64, CH3 / 64, 2), 256, 0, stream>>>(
      w_qkv, x, qkv_raw, CH3, DIMC, NTOT);

  // 2) depthwise 3x3
  const long total = (long)qkv_elems;
  dwconv3x3<<<(unsigned)((total + 255) / 256), 256, 0, stream>>>(
      qkv_raw, w_dw, qkv_dw, total, CH3);

  // 3) inverse L2 norms for q and k rows
  l2norm_rows<<<2 * 768, 256, 0, stream>>>(qkv_dw, scale);

  // 4) attn = temp * (q_n k_n^T)   via WMMA, K=16384 split over 8 waves
  attn_gram<<<dim3(9, 2 * NHEAD), 256, 0, stream>>>(qkv_dw, scale, temp, attn);

  // 5) combined top-k softmax weights P = sum_i a_i * p_i
  topk_combine<<<3, 256, 0, stream>>>(attn, P, a1, a2, a3, a4);

  // 6) attn_out = P @ v (block-diagonal over heads)
  attn_apply<<<dim3(NTOT / 128, 2 * NHEAD), 256, 0, stream>>>(P, qkv_dw, attn_out);

  // 7) out = w_proj @ attn_out   [384x384]x[384x16384] per batch
  gemm_wmma_f16<<<dim3(NTOT / 64, DIMC / 64, 2), 256, 0, stream>>>(
      w_proj, attn_out, outp, DIMC, DIMC, NTOT);
}
